// SelfAttention_13408887899041
// MI455X (gfx1250) — matrix operand order repaired and checked by reading, not compiled, not run on hardware
//
#include <hip/hip_runtime.h>
#include <hip/hip_bf16.h>
#include <stdint.h>
#include <stddef.h>

// ---------------------------------------------------------------------------
// Self-attention, B=2 N=2048 DIM=1024 H=16 D=64, fp32 in/out.
// Compute-bound (~60 GFLOP over ~44MB): all matmuls via v_wmma_f32_16x16x32_bf16
// (fp32 accumulate). Softmax scale 1/8 folded into Q. Shared operands (W tiles,
// K/V blocks) staged workgroup-wide into LDS with async-to-LDS DMA, double
// buffered on ASYNCcnt.
// ---------------------------------------------------------------------------

typedef __attribute__((ext_vector_type(16))) __bf16 v16bf;
typedef __attribute__((ext_vector_type(8)))  __bf16 bf16x8;
typedef __attribute__((ext_vector_type(4)))  __bf16 bf16x4;
typedef __attribute__((ext_vector_type(8)))  float  v8f;
typedef int v4i_vs __attribute__((vector_size(16)));   // builtin's pointee type

#define B_   2
#define N_   2048
#define DIM_ 1024
#define H_   16
#define D_   64

// ---- CDNA5 async-to-LDS plumbing (guarded; falls back to sync copies) -----
#if defined(__AMDGCN__) && __has_builtin(__builtin_amdgcn_global_load_async_to_lds_b128)
#define ASYNC_LDS 1
#else
#define ASYNC_LDS 0
#endif

__device__ __forceinline__ void cp16_lds(void* dst_lds, const void* src_glob) {
#if ASYNC_LDS
    // Builtin signature: (v4i addrspace(1)*, v4i addrspace(3)*, imm, imm).
    // Cast through integers: generic->AS1 keeps the 64-bit global address;
    // generic->AS3 truncates to addr[31:0] which IS the LDS offset (ISA 10.2).
    __builtin_amdgcn_global_load_async_to_lds_b128(
        (__attribute__((address_space(1))) v4i_vs*)(uintptr_t)src_glob,
        (__attribute__((address_space(3))) v4i_vs*)(uint32_t)(uintptr_t)dst_lds,
        0, 0);
#else
    *(bf16x8*)dst_lds = *(const bf16x8*)src_glob;
#endif
}

#if ASYNC_LDS
#if __has_builtin(__builtin_amdgcn_s_wait_asynccnt)
#define WAIT_ASYNC(n) __builtin_amdgcn_s_wait_asynccnt(n)
#else
#define WAIT_ASYNC(n) asm volatile("s_wait_asynccnt %0" ::"i"(n) : "memory")
#endif
#else
#define WAIT_ASYNC(n) ((void)0)
#endif

// ---------------------------------------------------------------------------
__device__ __forceinline__ __bf16 f2bf(float f) {
    uint32_t u = __builtin_bit_cast(uint32_t, f);
    uint32_t r = u + 0x7FFFu + ((u >> 16) & 1u);     // round-to-nearest-even
    uint16_t h = (uint16_t)(r >> 16);
    return __builtin_bit_cast(__bf16, h);
}

union Frag { bf16x8 h[2]; v16bf v; };

// A-matrix 16x32 bf16 fragment (ISA 7.12.2): lane L -> row L%16,
// K = {0..7,16..23} for lanes 0-15, {8..15,24..31} for lanes 16-31.
__device__ __forceinline__ v16bf load_a(const __bf16* base, int stride, int lane) {
    const __bf16* p = base + (size_t)(lane & 15) * stride + ((lane >> 4) << 3);
    Frag f;
    f.h[0] = *(const bf16x8*)(p);
    f.h[1] = *(const bf16x8*)(p + 16);
    return f.v;
}

// B-matrix 32x16 bf16 fragment: lane L -> column L%16,
// K = {0..7,16..23} for lanes 0-15, {8..15,24..31} for lanes 16-31. Element (k,n) at base + n*stride + k.
__device__ __forceinline__ v16bf load_b(const __bf16* base, int stride, int lane) {
    const __bf16* p = base + (size_t)(lane & 15) * stride + ((lane >> 4) << 3);
    Frag f;
    f.h[0] = *(const bf16x8*)(p);
    f.h[1] = *(const bf16x8*)(p + 16);
    return f.v;
}

__device__ __forceinline__ v8f wmma_bf16(v16bf a, v16bf b, v8f c) {
    return __builtin_amdgcn_wmma_f32_16x16x32_bf16(false, a, false, b, (short)0, c, false, false);
}

__device__ __forceinline__ float rmax16(float v) {
    #pragma unroll
    for (int m = 1; m < 16; m <<= 1) v = fmaxf(v, __shfl_xor(v, m, 32));
    return v;
}
__device__ __forceinline__ float rsum16(float v) {
    #pragma unroll
    for (int m = 1; m < 16; m <<= 1) v += __shfl_xor(v, m, 32);
    return v;
}

// ---------------------------------------------------------------------------
// Kernel 1: fp32 -> bf16 cast, 4 elements/thread.
// ---------------------------------------------------------------------------
__global__ void cast_f32_bf16(const float* __restrict__ src,
                              __bf16* __restrict__ dst, int n4) {
    int i = blockIdx.x * blockDim.x + threadIdx.x;
    if (i < n4) {
        float4 f = ((const float4*)src)[i];
        bf16x4 o;
        o.x = f2bf(f.x); o.y = f2bf(f.y); o.z = f2bf(f.z); o.w = f2bf(f.w);
        ((bf16x4*)dst)[i] = o;
    }
}

// ---------------------------------------------------------------------------
// Kernel 2: QKV projection. qkv[m,e] = sum_k x[m,k] * W[e,k].
// Block = 8 waves, each wave a 16(M)x64(E) tile; all waves share the same
// E-tile, so the 64x32 W k-slab is staged once per block into LDS (async DMA,
// double buffered). A fragments are prefetched one k-step ahead.
// Epilogue scatters to:  Q:[B,H,N,D] bf16 (pre-scaled 1/8), K:[B,H,N,D] bf16,
//                        V:[B,H,D,N] bf16 (transposed for P*V B-fragments).
// ---------------------------------------------------------------------------
__global__ __launch_bounds__(256) void qkv_gemm(const __bf16* __restrict__ xb,
                                                const __bf16* __restrict__ wb,
                                                __bf16* __restrict__ qb,
                                                __bf16* __restrict__ kb,
                                                __bf16* __restrict__ vb) {
    __shared__ alignas(16) __bf16 wslab[2][64 * 32];   // [col][k], 4KB each

    const int tid  = threadIdx.x;
    const int lane = tid & 31;
    const int wid  = tid >> 5;
    const int m0 = (blockIdx.y * 8 + wid) * 16;   // token-row tile
    const int e0 = blockIdx.x * 64;               // output-col tile (one head)

    // W-slab copy assignment: 256 threads x one 16B chunk.
    const int wcol = tid >> 2, wsub = tid & 3;    // col 0..63, 8-elt chunk 0..3
    const __bf16* wsrc = wb + (size_t)(e0 + wcol) * DIM_ + wsub * 8;
    __bf16* wdst0 = &wslab[0][wcol * 32 + wsub * 8];
    __bf16* wdst1 = &wslab[1][wcol * 32 + wsub * 8];

    v8f acc[4] = {v8f{}, v8f{}, v8f{}, v8f{}};
    const __bf16* arow = xb + (size_t)m0 * DIM_;

    cp16_lds(wdst0, wsrc);                        // slab for k-step 0
    v16bf a_cur = load_a(arow, DIM_, lane);

    const int STEPS = DIM_ / 32;                  // 32
    for (int s = 0; s < STEPS; ++s) {
        const int buf = s & 1;
        v16bf a_next;
        if (s + 1 < STEPS) {
            cp16_lds(buf ? wdst0 : wdst1, wsrc + (s + 1) * 32);
            a_next = load_a(arow + (s + 1) * 32, DIM_, lane);
            WAIT_ASYNC(1);                        // slab s done, s+1 in flight
        } else {
            WAIT_ASYNC(0);
        }
        __syncthreads();
        #pragma unroll
        for (int c = 0; c < 4; ++c) {
            v16bf b = load_b(&wslab[buf][(c * 16) * 32], 32, lane);
            acc[c] = wmma_bf16(a_cur, b, acc[c]);
        }
        __syncthreads();                          // slab free for next DMA
        if (s + 1 < STEPS) a_cur = a_next;
    }

    const int col   = lane & 15;
    const int half8 = (lane >> 4) << 3;           // C rows: i + (lane>=16 ? 8:0)
    const int seg   = e0 >> 10;                   // 0=Q 1=K 2=V
    const int h     = (e0 & 1023) >> 6;

    #pragma unroll
    for (int c = 0; c < 4; ++c) {
        const int d = c * 16 + col;
        #pragma unroll
        for (int i = 0; i < 8; ++i) {
            const int row = m0 + i + half8;
            const int b   = row >> 11;
            const int n   = row & (N_ - 1);
            const float v = acc[c][i];
            const size_t bh = (size_t)(b * H_ + h);
            if (seg == 0)      qb[((bh * N_ + n) << 6) + d] = f2bf(v * 0.125f);
            else if (seg == 1) kb[((bh * N_ + n) << 6) + d] = f2bf(v);
            else               vb[((bh * D_ + d) << 11) + n] = f2bf(v);
        }
    }
}

// ---------------------------------------------------------------------------
// Kernel 3: flash attention. Block = 4 waves x 16 query rows = 64 rows; all
// waves share the same (b,h), so each 32-key K block (32x64) and V block
// (64x32, [d][j]) is staged once per block into LDS via async DMA, double
// buffered (4 async instrs/step -> s_wait_asynccnt 4 overlaps next DMA).
// ---------------------------------------------------------------------------
__global__ __launch_bounds__(128) void attn_kernel(const __bf16* __restrict__ qb,
                                                   const __bf16* __restrict__ kb,
                                                   const __bf16* __restrict__ vb,
                                                   float* __restrict__ out) {
    __shared__ alignas(16) __bf16 kslab[2][32 * 64];   // [j][d], 4KB each
    __shared__ alignas(16) __bf16 vslab[2][64 * 32];   // [d][j], 4KB each
    __shared__ alignas(16) __bf16 pbuf[4][16 * 32];    // per-wave P staging

    const int tid  = threadIdx.x;
    const int lane = tid & 31;
    const int wid  = tid >> 5;
    const int bh   = blockIdx.y;                       // b*H + h
    const int r0   = blockIdx.x * 64 + wid * 16;       // query-row tile

    const __bf16* kg = kb + (size_t)bh * N_ * D_;
    const __bf16* vg = vb + (size_t)bh * D_ * N_;

    const __bf16* qrow = qb + ((size_t)bh * N_ + r0) * D_;
    const v16bf qa0 = load_a(qrow,      D_, lane);     // d = 0..31
    const v16bf qa1 = load_a(qrow + 32, D_, lane);     // d = 32..63

    float m_i[8], l_i[8];
    #pragma unroll
    for (int i = 0; i < 8; ++i) { m_i[i] = -1e30f; l_i[i] = 0.0f; }
    v8f oacc[4] = {v8f{}, v8f{}, v8f{}, v8f{}};

    __bf16* pl = &pbuf[wid][0];
    const int col = lane & 15;
    const int kh  = lane >> 4;

    // K/V copy assignment: 256 16B chunks each, 128 threads -> 2 chunks each.
    // K chunk c: row j=c>>3 (64 elts = 8 chunks), V chunk c: row d=c>>2 (4 chunks)
    auto issue_step = [&](int j0, int buf) {
        #pragma unroll
        for (int r = 0; r < 2; ++r) {
            const int c  = tid + r * 128;
            const int kj = c >> 3, ks = c & 7;
            cp16_lds(&kslab[buf][kj * 64 + ks * 8],
                     kg + (size_t)(j0 + kj) * D_ + ks * 8);
            const int vd = c >> 2, vs = c & 3;
            cp16_lds(&vslab[buf][vd * 32 + vs * 8],
                     vg + (size_t)vd * N_ + j0 + vs * 8);
        }
    };

    const int STEPS = N_ / 32;                         // 64
    issue_step(0, 0);
    for (int s = 0; s < STEPS; ++s) {
        const int buf = s & 1;
        if (s + 1 < STEPS) {
            issue_step((s + 1) * 32, buf ^ 1);
            WAIT_ASYNC(4);                             // step s done, s+1 in flight
        } else {
            WAIT_ASYNC(0);
        }
        __syncthreads();

        const __bf16* kk = &kslab[buf][0];
        const v8f z = {};
        v8f s0 = wmma_bf16(qa0, load_b(kk,                D_, lane), z);
        s0     = wmma_bf16(qa1, load_b(kk + 32,           D_, lane), s0);
        v8f s1 = wmma_bf16(qa0, load_b(kk + 16 * D_,      D_, lane), z);
        s1     = wmma_bf16(qa1, load_b(kk + 16 * D_ + 32, D_, lane), s1);

        float p0[8], p1[8], alpha[8];
        #pragma unroll
        for (int i = 0; i < 8; ++i) {
            float mx = rmax16(fmaxf(s0[i], s1[i]));
            float mn = fmaxf(m_i[i], mx);
            p0[i] = __expf(s0[i] - mn);
            p1[i] = __expf(s1[i] - mn);
            float rs = rsum16(p0[i] + p1[i]);
            alpha[i] = __expf(m_i[i] - mn);
            l_i[i] = l_i[i] * alpha[i] + rs;
            m_i[i] = mn;
        }
        #pragma unroll
        for (int c = 0; c < 4; ++c)
            #pragma unroll
            for (int i = 0; i < 8; ++i) oacc[c][i] *= alpha[i];

        // C-layout -> A-layout via per-wave LDS staging (same wave, in-order DS)
        #pragma unroll
        for (int i = 0; i < 8; ++i) {
            const int row = i + (kh << 3);
            pl[row * 32 + col]      = f2bf(p0[i]);
            pl[row * 32 + 16 + col] = f2bf(p1[i]);
        }
        const v16bf pa = load_a(pl, 32, lane);

        const __bf16* vv = &vslab[buf][0];
        #pragma unroll
        for (int c = 0; c < 4; ++c) {
            v16bf vf = load_b(vv + (c * 16) * 32, 32, lane);
            oacc[c] = wmma_bf16(pa, vf, oacc[c]);
        }
        __syncthreads();                               // slabs free for next DMA
    }

    // Normalize and store: out[b][n*H + h][d] (fp32)
    const int b = bh >> 4, h = bh & 15;
    #pragma unroll
    for (int i = 0; i < 8; ++i) {
        const float inv = 1.0f / l_i[i];
        const int n = r0 + i + (kh << 3);
        #pragma unroll
        for (int c = 0; c < 4; ++c) {
            const int d = c * 16 + col;
            out[(((size_t)(b * N_ + n) * H_ + h) << 6) + d] = oacc[c][i] * inv;
        }
    }
}

// ---------------------------------------------------------------------------
extern "C" void kernel_launch(void* const* d_in, const int* in_sizes, int n_in,
                              void* d_out, int out_size, void* d_ws, size_t ws_size,
                              hipStream_t stream) {
    const float* x = (const float*)d_in[0];   // [B, N, DIM]
    const float* w = (const float*)d_in[1];   // [3*DIM, DIM]
    float* out = (float*)d_out;               // [B, N*H, D]

    char* ws = (char*)d_ws;
    __bf16* xb = (__bf16*)(ws);                          //  8 MB: [B*N, DIM]
    __bf16* wb = (__bf16*)(ws + (size_t)8  * 1048576);   //  6 MB: [3*DIM, DIM]
    __bf16* qb = (__bf16*)(ws + (size_t)14 * 1048576);   //  8 MB: [B,H,N,D] (pre-scaled)
    __bf16* kb = (__bf16*)(ws + (size_t)22 * 1048576);   //  8 MB: [B,H,N,D]
    __bf16* vb = (__bf16*)(ws + (size_t)30 * 1048576);   //  8 MB: [B,H,D,N]

    const int nx4 = (B_ * N_ * DIM_) / 4;     // 1,048,576
    const int nw4 = (3 * DIM_ * DIM_) / 4;    //   786,432
    cast_f32_bf16<<<(nx4 + 255) / 256, 256, 0, stream>>>(x, xb, nx4);
    cast_f32_bf16<<<(nw4 + 255) / 256, 256, 0, stream>>>(w, wb, nw4);

    // GEMM: 3072/64 = 48 col tiles, 4096/(16 rows * 8 waves) = 32 row tiles
    qkv_gemm<<<dim3(48, 32), 256, 0, stream>>>(xb, wb, qb, kb, vb);

    // Attention: 2048/64 = 32 query blocks x (B*H = 32) heads
    attn_kernel<<<dim3(32, 32), 128, 0, stream>>>(qb, kb, vb, out);
}
